// MemoryModule_46797963657834
// MI455X (gfx1250) — matrix operand order repaired and checked by reading, not compiled, not run on hardware
//
#include <hip/hip_runtime.h>
#include <hip/hip_bf16.h>

typedef __bf16 bf16;
typedef __attribute__((ext_vector_type(16))) __bf16 v16bf;
typedef __attribute__((ext_vector_type(8)))  __bf16 v8bf;
typedef __attribute__((ext_vector_type(8)))  float  v8f;

#define R_TOT 8192   // B*S rows
#define H_DIM 1024
#define M_DIM 8192

__device__ __forceinline__ v16bf cat8(v8bf lo, v8bf hi) {
  v16bf r;
#pragma unroll
  for (int i = 0; i < 8; ++i) { r[i] = lo[i]; r[i + 8] = hi[i]; }
  return r;
}

__device__ __forceinline__ v8f wmma_bf16(v16bf a, v16bf b, v8f c) {
  // D = A(16x32 bf16) * B(32x16 bf16) + C(16x16 f32)
  return __builtin_amdgcn_wmma_f32_16x16x32_bf16(false, a, false, b, (short)0, c,
                                                 false, false);
}

// CDNA5 async global->LDS copy (ASYNCcnt-tracked), SGPR-base + 32-bit lane
// offset (GVS form) to keep per-lane address state to a single VGPR.
__device__ __forceinline__ void async_ld_b128_s(unsigned lds_off, unsigned voff,
                                                const void* sbase) {
  asm volatile("global_load_async_to_lds_b128 %0, %1, %2"
               :: "v"(lds_off), "v"(voff), "s"(sbase) : "memory");
}
__device__ __forceinline__ void wait_async0() {
  asm volatile("s_wait_asynccnt 0x0" ::: "memory");
}

// ---------------- small utility kernels ----------------
__global__ __launch_bounds__(256) void k_zero(float* __restrict__ p, int n) {
  int i = blockIdx.x * 256 + threadIdx.x;
  if (i < n) p[i] = 0.f;
}

__global__ __launch_bounds__(256) void k_cvt(const float* __restrict__ s,
                                             bf16* __restrict__ d, int n) {
  int i = (blockIdx.x * 256 + threadIdx.x) * 4;
  if (i + 3 < n) {
    float4 v = *(const float4*)(s + i);
    d[i + 0] = (bf16)v.x; d[i + 1] = (bf16)v.y;
    d[i + 2] = (bf16)v.z; d[i + 3] = (bf16)v.w;
  }
}

// ---------------- logits + exp + row-sum ----------------
// P[s,m] = exp( x[s,:] . w[m,:] + bias[m] ),  rowsum[s] += sum_m P[s,m]
// Workgroup: 128 rows x 128-m tiles; 8 waves as 4(row)x2(col); wave tile 32x64
// (8 accumulators, 16 WMMAs per 64-wide K stage). Double-buffered async
// global->LDS staging overlaps DMA with WMMA.
__global__ __launch_bounds__(256) void k_logits(
    const bf16* __restrict__ xb, const bf16* __restrict__ wb,
    const float* __restrict__ bias, bf16* __restrict__ P,
    float* __restrict__ rowsum) {
  __shared__ bf16 xs[2][128][64];   // [buf][row][k]
  __shared__ bf16 ws[2][128][64];   // [buf][m-in-tile][k]

  const int tid  = threadIdx.x;
  const int wv   = tid >> 5, lane = tid & 31;
  const int half = lane >> 4, ln = lane & 15;
  const int rowblk = blockIdx.x * 128;
  const int wr = wv & 3;        // wave row group (32 rows each)
  const int wc = wv >> 2;       // wave col group (64 cols each)

  const int sr = tid >> 1, sc = (tid & 1) * 32;  // staging coords
  const unsigned lx[2] = {(unsigned)(uintptr_t)&xs[0][sr][sc],
                          (unsigned)(uintptr_t)&xs[1][sr][sc]};
  const unsigned lw[2] = {(unsigned)(uintptr_t)&ws[0][sr][sc],
                          (unsigned)(uintptr_t)&ws[1][sr][sc]};
  // per-lane byte offset within a [128][H_DIM-strided] source tile
  const unsigned vo = (unsigned)(sr * H_DIM + sc) * 2u;

  auto issue = [&](int buf, const bf16* bx, const bf16* bw) {
#pragma unroll
    for (int j = 0; j < 4; ++j) {
      async_ld_b128_s(lx[buf] + 16u * j, vo + 16u * j, bx);
      async_ld_b128_s(lw[buf] + 16u * j, vo + 16u * j, bw);
    }
  };

  for (int m0 = 0; m0 < M_DIM; m0 += 128) {
    v8f acc[2][4] = {};
    issue(0, xb + (size_t)rowblk * H_DIM, wb + (size_t)m0 * H_DIM);
    int buf = 0;
    for (int k0 = 0; k0 < H_DIM; k0 += 64, buf ^= 1) {
      wait_async0();          // this wave's stage-k writes done
      __syncthreads();        // everyone's writes done, prev reads done
      if (k0 + 64 < H_DIM)    // DMA for stage k+1 under this stage's WMMAs
        issue(buf ^ 1, xb + (size_t)rowblk * H_DIM + k0 + 64,
              wb + (size_t)m0 * H_DIM + k0 + 64);
#pragma unroll
      for (int s2 = 0; s2 < 64; s2 += 32) {
        v16bf a[2];
#pragma unroll
        for (int rt = 0; rt < 2; ++rt) {
          const bf16* p = &xs[buf][32 * wr + 16 * rt + ln][s2 + half * 8];
          a[rt] = cat8(*(const v8bf*)p, *(const v8bf*)(p + 16));
        }
#pragma unroll
        for (int ct = 0; ct < 4; ++ct) {
          const bf16* p = &ws[buf][64 * wc + 16 * ct + ln][s2 + half * 8];
          v16bf b = cat8(*(const v8bf*)p, *(const v8bf*)(p + 16));
#pragma unroll
          for (int rt = 0; rt < 2; ++rt)
            acc[rt][ct] = wmma_bf16(a[rt], b, acc[rt][ct]);
        }
      }
    }
    // epilogue: bias, exp, store bf16 weights, reduce row sums
#pragma unroll
    for (int rt = 0; rt < 2; ++rt) {
      float rs[8] = {0.f, 0.f, 0.f, 0.f, 0.f, 0.f, 0.f, 0.f};
#pragma unroll
      for (int ct = 0; ct < 4; ++ct) {
        int mcol = m0 + 64 * wc + 16 * ct + ln;
        float bv = bias[mcol];
#pragma unroll
        for (int e = 0; e < 8; ++e) {
          float ex = __expf(acc[rt][ct][e] + bv);
          int srow = rowblk + 32 * wr + 16 * rt + 8 * half + e;
          P[(size_t)srow * M_DIM + mcol] = (bf16)ex;
          rs[e] += ex;
        }
      }
#pragma unroll
      for (int e = 0; e < 8; ++e) {
        float v = rs[e];
#pragma unroll
        for (int sh = 1; sh < 16; sh <<= 1) v += __shfl_xor(v, sh, 32);
        if (ln == 0)
          atomicAdd(&rowsum[rowblk + 32 * wr + 16 * rt + 8 * half + e], v);
      }
    }
  }
}

// ---------------- read path: O[s,h] = (1/rowsum[s]) * sum_m P[s,m]*mem[m,h]
// Register-pipelined transpose staging: next chunk's global loads issue under
// the current chunk's WMMAs.
__global__ __launch_bounds__(256) void k_read(
    const bf16* __restrict__ P, const float* __restrict__ rowsum,
    const bf16* __restrict__ memb, float* __restrict__ outR) {
  __shared__ bf16 bt[128][64];   // [h-in-tile][k] (transposed memory tile)
  const int tid  = threadIdx.x;
  const int wv   = tid >> 5, lane = tid & 31;
  const int half = lane >> 4, ln = lane & 15;
  const int rowblk = blockIdx.x * 128;
  const int hbase  = blockIdx.y * 128;
  const int wr = wv & 3, wc = wv >> 2;

  const int kk = tid >> 2, c0 = (tid & 3) * 32;
  v8bf rv[4];
  {
    const bf16* g = memb + (size_t)kk * H_DIM + hbase + c0;
#pragma unroll
    for (int j = 0; j < 4; ++j) rv[j] = *(const v8bf*)(g + 8 * j);
  }

  v8f acc[2][4] = {};
  for (int k0 = 0; k0 < M_DIM; k0 += 64) {
    __syncthreads();   // prev compute done reading bt
#pragma unroll
    for (int j = 0; j < 4; ++j)
#pragma unroll
      for (int i = 0; i < 8; ++i) bt[c0 + 8 * j + i][kk] = rv[j][i];
    __syncthreads();   // bt ready
    if (k0 + 64 < M_DIM) {  // pipeline next chunk under compute
      const bf16* g = memb + (size_t)(k0 + 64 + kk) * H_DIM + hbase + c0;
#pragma unroll
      for (int j = 0; j < 4; ++j) rv[j] = *(const v8bf*)(g + 8 * j);
    }
#pragma unroll
    for (int s2 = 0; s2 < 64; s2 += 32) {
      v16bf a[2];
#pragma unroll
      for (int rt = 0; rt < 2; ++rt) {
        const bf16* ga =
            P + (size_t)(rowblk + 32 * wr + 16 * rt + ln) * M_DIM + k0 + s2;
        a[rt] = cat8(*(const v8bf*)(ga + half * 8),
                     *(const v8bf*)(ga + 16 + half * 8));
      }
#pragma unroll
      for (int ct = 0; ct < 4; ++ct) {
        const bf16* p = &bt[64 * wc + 16 * ct + ln][s2 + half * 8];
        v16bf b = cat8(*(const v8bf*)p, *(const v8bf*)(p + 16));
#pragma unroll
        for (int rt = 0; rt < 2; ++rt)
          acc[rt][ct] = wmma_bf16(a[rt], b, acc[rt][ct]);
      }
    }
  }
#pragma unroll
  for (int rt = 0; rt < 2; ++rt) {
#pragma unroll
    for (int e = 0; e < 8; ++e) {
      int srow = rowblk + 32 * wr + 16 * rt + 8 * half + e;
      float inv = 1.0f / rowsum[srow];
#pragma unroll
      for (int ct = 0; ct < 4; ++ct)
        outR[(size_t)srow * H_DIM + hbase + 64 * wc + 16 * ct + ln] =
            acc[rt][ct][e] * inv;
    }
  }
}

// ---------------- write path: newmem[m,h] = mem[m,h] + sum_s (P[s,m]/rowsum[s])*x[s,h]
__global__ __launch_bounds__(256) void k_write(
    const bf16* __restrict__ P, const float* __restrict__ rowsum,
    const bf16* __restrict__ xb, const float* __restrict__ memory,
    float* __restrict__ outM) {
  __shared__ bf16 at[128][64];   // [m-in-tile][s] (transposed, pre-normalized P)
  __shared__ bf16 xt[128][64];   // [h-in-tile][s] (transposed x)
  const int tid  = threadIdx.x;
  const int wv   = tid >> 5, lane = tid & 31;
  const int half = lane >> 4, ln = lane & 15;
  const int mbase = blockIdx.x * 128;
  const int hbase = blockIdx.y * 128;
  const int wr = wv & 3, wc = wv >> 2;

  const int ss = tid >> 2, c0 = (tid & 3) * 32;
  v8bf rvp[4], rvx[4];
  float rsum_cur;
  {
    const bf16* gp = P + (size_t)ss * M_DIM + mbase + c0;
    const bf16* gx = xb + (size_t)ss * H_DIM + hbase + c0;
#pragma unroll
    for (int j = 0; j < 4; ++j) {
      rvp[j] = *(const v8bf*)(gp + 8 * j);
      rvx[j] = *(const v8bf*)(gx + 8 * j);
    }
    rsum_cur = rowsum[ss];
  }

  v8f acc[2][4] = {};
  for (int k0 = 0; k0 < R_TOT; k0 += 64) {
    __syncthreads();
    float inv = 1.0f / rsum_cur;
#pragma unroll
    for (int j = 0; j < 4; ++j)
#pragma unroll
      for (int i = 0; i < 8; ++i) {
        at[c0 + 8 * j + i][ss] = (bf16)((float)rvp[j][i] * inv);
        xt[c0 + 8 * j + i][ss] = rvx[j][i];
      }
    __syncthreads();
    if (k0 + 64 < R_TOT) {  // pipeline next chunk under compute
      const bf16* gp = P + (size_t)(k0 + 64 + ss) * M_DIM + mbase + c0;
      const bf16* gx = xb + (size_t)(k0 + 64 + ss) * H_DIM + hbase + c0;
#pragma unroll
      for (int j = 0; j < 4; ++j) {
        rvp[j] = *(const v8bf*)(gp + 8 * j);
        rvx[j] = *(const v8bf*)(gx + 8 * j);
      }
      rsum_cur = rowsum[k0 + 64 + ss];
    }
#pragma unroll
    for (int s2 = 0; s2 < 64; s2 += 32) {
      v16bf a[2];
#pragma unroll
      for (int rt = 0; rt < 2; ++rt) {
        const bf16* p = &at[32 * wr + 16 * rt + ln][s2 + half * 8];
        a[rt] = cat8(*(const v8bf*)p, *(const v8bf*)(p + 16));
      }
#pragma unroll
      for (int ct = 0; ct < 4; ++ct) {
        const bf16* p = &xt[64 * wc + 16 * ct + ln][s2 + half * 8];
        v16bf b = cat8(*(const v8bf*)p, *(const v8bf*)(p + 16));
#pragma unroll
        for (int rt = 0; rt < 2; ++rt)
          acc[rt][ct] = wmma_bf16(a[rt], b, acc[rt][ct]);
      }
    }
  }
#pragma unroll
  for (int rt = 0; rt < 2; ++rt) {
#pragma unroll
    for (int e = 0; e < 8; ++e) {
      int m = mbase + 32 * wr + 16 * rt + 8 * half + e;
#pragma unroll
      for (int ct = 0; ct < 4; ++ct) {
        size_t i0 = (size_t)m * H_DIM + hbase + 64 * wc + 16 * ct + ln;
        outM[i0] = memory[i0] + acc[rt][ct][e];
      }
    }
  }
}

// ---------------- host launcher ----------------
extern "C" void kernel_launch(void* const* d_in, const int* in_sizes, int n_in,
                              void* d_out, int out_size, void* d_ws, size_t ws_size,
                              hipStream_t stream) {
  const float* x       = (const float*)d_in[0];
  const float* memory  = (const float*)d_in[1];
  const float* read_w  = (const float*)d_in[2];
  const float* read_b  = (const float*)d_in[3];
  const float* write_w = (const float*)d_in[4];
  const float* write_b = (const float*)d_in[5];
  float* out = (float*)d_out;
  char* ws = (char*)d_ws;

  const size_t RH = (size_t)R_TOT * H_DIM;
  const size_t MH = (size_t)M_DIM * H_DIM;

  // workspace layout (bytes):
  //   [0,16M)    x bf16          [16M,32M)  read_w bf16
  //   [32M,48M)  write_w bf16    [48M,64M)  memory bf16
  //   [64M,192M) P (exp weights, bf16, 8192x8192)
  //   [192M,+32K) rowsum f32
  bf16* xb      = (bf16*)(ws);
  bf16* rwb     = (bf16*)(ws + (size_t)16 * 1024 * 1024);
  bf16* wwb     = (bf16*)(ws + (size_t)32 * 1024 * 1024);
  bf16* memb    = (bf16*)(ws + (size_t)48 * 1024 * 1024);
  bf16* P       = (bf16*)(ws + (size_t)64 * 1024 * 1024);
  float* rowsum = (float*)(ws + (size_t)192 * 1024 * 1024);

  dim3 blk(256);
  k_cvt<<<(unsigned)((RH / 4 + 255) / 256), blk, 0, stream>>>(x, xb, (int)RH);
  k_cvt<<<(unsigned)((MH / 4 + 255) / 256), blk, 0, stream>>>(read_w, rwb, (int)MH);
  k_cvt<<<(unsigned)((MH / 4 + 255) / 256), blk, 0, stream>>>(write_w, wwb, (int)MH);
  k_cvt<<<(unsigned)((MH / 4 + 255) / 256), blk, 0, stream>>>(memory, memb, (int)MH);

  // read path
  k_zero<<<(R_TOT + 255) / 256, blk, 0, stream>>>(rowsum, R_TOT);
  k_logits<<<R_TOT / 128, blk, 0, stream>>>(xb, rwb, read_b, P, rowsum);
  k_read<<<dim3(R_TOT / 128, H_DIM / 128), blk, 0, stream>>>(P, rowsum, memb, out);

  // write path (reuses P / rowsum scratch after read path completes in-stream)
  k_zero<<<(R_TOT + 255) / 256, blk, 0, stream>>>(rowsum, R_TOT);
  k_logits<<<R_TOT / 128, blk, 0, stream>>>(xb, wwb, write_b, P, rowsum);
  k_write<<<dim3(M_DIM / 128, H_DIM / 128), blk, 0, stream>>>(P, rowsum, xb, memory,
                                                              out + RH);
}
